// ViT_M_32023276159446
// MI455X (gfx1250) — compile-verified, matrix operand order checked by reading, not run on hardware
//
#include <hip/hip_runtime.h>
#include <hip/hip_bf16.h>
#include <math.h>

// ---------------------------------------------------------------------------
// ViT-Base forward for MI455X (gfx1250, wave32, WMMA 16x16x32 f16)
// GEMM: double-buffered LDS fed by global_load_async_to_lds_b128 (ASYNCcnt),
// conflict-free padded LDS strides, 128x128 tile, 8 WMMAs per wave per K-step.
// ---------------------------------------------------------------------------

#define D_MODEL 768
#define NHEAD   12
#define HDIM    64
#define MLPDIM  3072
#define NLAYER  12
#define PATCH   16
#define IMG     224
#define NX      14
#define NT      196
#define TSEQ    197
#define BATCH   32
#define NROWS   (BATCH * TSEQ)   // 6304
#define NROWS1  (BATCH * NT)     // 6272
#define OUTC    256
#define EPSLN   1e-5f
#define TPAD    224              // key dim padded to 7*32

// padded LDS K-strides (halfs) for conflict-free 16B fragment reads
#define LDK 40    // gemm A/B tiles (rows of 32 used)
#define QLD 72    // attention Q/K tiles (rows of 64 used)
#define PLD 232   // attention P tile (rows of 224 used)
#define SLD 225   // attention scores (f32)
#define VLD 40    // attention V tile (rows of 32 used)

typedef __attribute__((ext_vector_type(16))) _Float16 v16h;
typedef __attribute__((ext_vector_type(8)))  _Float16 v8h;
typedef __attribute__((ext_vector_type(8)))  float    v8f;

// Load one 16x32 f16 fragment (A layout; B uses same layout on a transposed
// [n][k] LDS tile).  Per ISA 7.12.2: lane L holds row (L&15); elements 0..7
// are K = 8*(L>>4)..+7, elements 8..15 are K = 16+8*(L>>4)..+7.
__device__ __forceinline__ v16h ld_frag(const _Float16* tile, int ldk,
                                        int row0, int k0, int lane) {
  const _Float16* p = tile + (row0 + (lane & 15)) * ldk + k0 + 8 * (lane >> 4);
  v8h lo = *(const v8h*)(p);
  v8h hi = *(const v8h*)(p + 16);
  v16h r;
#pragma unroll
  for (int i = 0; i < 8; ++i) { r[i] = lo[i]; r[i + 8] = hi[i]; }
  return r;
}

__device__ __forceinline__ v8f wmma_f16(v16h a, v16h b, v8f c) {
  return __builtin_amdgcn_wmma_f32_16x16x32_f16(false, a, false, b,
                                                (short)0, c, false, false);
}

__device__ __forceinline__ float gelu_exact(float x) {
  return 0.5f * x * (1.0f + erff(x * 0.70710678118654752f));
}

// async 16B global -> LDS copy (per active lane); tracked by ASYNCcnt
__device__ __forceinline__ void async_b128(unsigned lds, const _Float16* g) {
  asm volatile("global_load_async_to_lds_b128 %0, %1, off"
               :: "v"(lds), "v"(g) : "memory");
}
__device__ __forceinline__ void wait_async0() {
  asm volatile("s_wait_asynccnt 0x0" ::: "memory");
}

// ---------------------------------------------------------------------------
// Operand prep kernels
// ---------------------------------------------------------------------------
__global__ __launch_bounds__(256)
void cvt_f16(const float* __restrict__ x, _Float16* __restrict__ y, long n8) {
  long i = (long)blockIdx.x * 256 + threadIdx.x;
  if (i >= n8) return;
  const float4* p = (const float4*)x + i * 2;
  float4 a = p[0], b = p[1];
  v8h h;
  h[0] = (_Float16)a.x; h[1] = (_Float16)a.y;
  h[2] = (_Float16)a.z; h[3] = (_Float16)a.w;
  h[4] = (_Float16)b.x; h[5] = (_Float16)b.y;
  h[6] = (_Float16)b.z; h[7] = (_Float16)b.w;
  *(v8h*)(y + i * 8) = h;
}

__global__ __launch_bounds__(256)
void zero_f16(_Float16* __restrict__ y, long n8) {
  long i = (long)blockIdx.x * 256 + threadIdx.x;
  if (i >= n8) return;
  v8h z = {};
  *(v8h*)(y + i * 8) = z;
}

// B f32 [K,N] -> Bt f16 [N,K] via LDS tile (K,N multiples of 32)
__global__ __launch_bounds__(256)
void transpose_f16(const float* __restrict__ B, _Float16* __restrict__ Bt,
                   int K, int N) {
  __shared__ float tile[32 * 33];
  const int k0 = blockIdx.y * 32, n0 = blockIdx.x * 32;
  for (int idx = threadIdx.x; idx < 1024; idx += 256) {
    int r = idx >> 5, c = idx & 31;
    tile[r * 33 + c] = B[(size_t)(k0 + r) * N + n0 + c];
  }
  __syncthreads();
  for (int idx = threadIdx.x; idx < 1024; idx += 256) {
    int rn = idx >> 5, ck = idx & 31;
    Bt[(size_t)(n0 + rn) * K + k0 + ck] = (_Float16)tile[ck * 33 + rn];
  }
}

// qkv (NROWS x 2304 f32, column(s,h,d) = d*36+s*12+h per reference reshape)
//  -> qb[bh][t][d], kb[bh][t][d], vb[bh][d][t'] f16  (t' padded to TPAD)
__global__ __launch_bounds__(256)
void permute_qkv(const float* __restrict__ qkv, _Float16* __restrict__ qb,
                 _Float16* __restrict__ kb, _Float16* __restrict__ vb) {
  size_t idx = (size_t)blockIdx.x * 256 + threadIdx.x;
  if (idx >= (size_t)NROWS * D_MODEL) return;
  int col = (int)(idx % D_MODEL);
  size_t row = idx / D_MODEL;
  int b = (int)(row / TSEQ), t = (int)(row % TSEQ);
  int h = col >> 6, d = col & 63;
  const float* src = qkv + row * (3 * D_MODEL) + d * 36 + h;
  size_t bh = (size_t)b * NHEAD + h;
  qb[(bh * TSEQ + t) * HDIM + d] = (_Float16)src[0];
  kb[(bh * TSEQ + t) * HDIM + d] = (_Float16)src[NHEAD];
  vb[(bh * HDIM + d) * TPAD + t] = (_Float16)src[2 * NHEAD];
}

// ---------------------------------------------------------------------------
// GEMM: C[M,N] = act(A16[M,K] @ Bt16[N,K]^T + bias)
// BM=BN=128, BK=32.  8 waves (2x4); wave tile 64x32 = 8 WMMAs per K-step.
// Double-buffered LDS staged via async global->LDS B128 copies.
// Requires K % 32 == 0, N % 128 == 0.  actmode: 0 none, 1 exact gelu.
// ---------------------------------------------------------------------------
__global__ __launch_bounds__(256)
void gemm_wmma(const _Float16* __restrict__ A16,
               const _Float16* __restrict__ Bt16,
               const float* __restrict__ bias, float* __restrict__ C,
               int M, int N, int K, int actmode) {
  __shared__ alignas(16) _Float16 As[2][128 * LDK];
  __shared__ alignas(16) _Float16 Bs[2][128 * LDK];

  const int tid  = threadIdx.x;
  const int lane = tid & 31;
  const int wave = tid >> 5;
  const int wm   = wave >> 2;          // 0..1 -> 64 rows each
  const int wn   = wave & 3;           // 0..3 -> 32 cols each
  const int mblk = blockIdx.y * 128;
  const int nblk = blockIdx.x * 128;

  // staging: each thread owns one row (tid>>1) x 16-half chunk ((tid&1)*16),
  // transferred as two async 16B copies.
  const int sr = tid >> 1;             // 0..127
  const int sc = (tid & 1) * 16;       // 0 or 16
  const bool avalid = (mblk + sr) < M;
  const _Float16* agp = A16  + (size_t)(mblk + sr) * K + sc;
  const _Float16* bgp = Bt16 + (size_t)(nblk + sr) * K + sc;
  const unsigned aofs = (unsigned)(size_t)&As[0][0] + (unsigned)(sr * LDK + sc) * 2u;
  const unsigned bofs = (unsigned)(size_t)&Bs[0][0] + (unsigned)(sr * LDK + sc) * 2u;
  const unsigned bufb = 128u * LDK * 2u;   // bytes per buffer

  v8f acc[4][2];
#pragma unroll
  for (int i = 0; i < 4; ++i)
#pragma unroll
    for (int j = 0; j < 2; ++j) acc[i][j] = {};

  auto stage = [&](int buf, int k0) {
    unsigned la = aofs + (unsigned)buf * bufb;
    unsigned lb = bofs + (unsigned)buf * bufb;
    const _Float16* ga = agp + k0;
    const _Float16* gb = bgp + k0;
    if (avalid) {
      async_b128(la,       ga);
      async_b128(la + 16u, ga + 8);
    }
    async_b128(lb,       gb);
    async_b128(lb + 16u, gb + 8);
  };

  const int nk = K / 32;
  stage(0, 0);
  for (int i = 0; i < nk; ++i) {
    const int cur = i & 1;
    wait_async0();           // my async stores into buffer `cur` done
    __syncthreads();         // everyone's done; prior reads of !cur done
    if (i + 1 < nk) stage(cur ^ 1, (i + 1) * 32);

    const _Float16* Ab = &As[cur][0];
    const _Float16* Bb = &Bs[cur][0];
    v16h bf[2], af[4];
#pragma unroll
    for (int j = 0; j < 2; ++j)
      bf[j] = ld_frag(Bb, LDK, wn * 32 + j * 16, 0, lane);
#pragma unroll
    for (int ii = 0; ii < 4; ++ii)
      af[ii] = ld_frag(Ab, LDK, wm * 64 + ii * 16, 0, lane);
#pragma unroll
    for (int ii = 0; ii < 4; ++ii)
#pragma unroll
      for (int j = 0; j < 2; ++j)
        acc[ii][j] = wmma_f16(af[ii], bf[j], acc[ii][j]);
  }

  const int half = lane >> 4;
  const int ln   = lane & 15;
#pragma unroll
  for (int ii = 0; ii < 4; ++ii)
#pragma unroll
    for (int j = 0; j < 2; ++j)
#pragma unroll
      for (int r = 0; r < 8; ++r) {
        int row = mblk + wm * 64 + ii * 16 + r + 8 * half;
        int col = nblk + wn * 32 + j * 16 + ln;
        if (row < M && col < N) {
          float v = acc[ii][j][r];
          if (bias) v += bias[col];
          if (actmode == 1) v = gelu_exact(v);
          C[(size_t)row * N + col] = v;
        }
      }
}

// ---------------------------------------------------------------------------
// Attention: one wave per (batch, head, 16-row q block).
// energy = (q@k^T) * sqrt(HD); softmax; o = attn@v -> attn_o[b,t,h*64+d]
// ---------------------------------------------------------------------------
__global__ __launch_bounds__(32)
void attn_wmma(const _Float16* __restrict__ qb, const _Float16* __restrict__ kb,
               const _Float16* __restrict__ vb, float* __restrict__ attn_o) {
  __shared__ alignas(16) float    S[16 * SLD];
  __shared__ alignas(16) _Float16 Ps[16 * PLD];
  __shared__ alignas(16) _Float16 Qs[16 * QLD];
  __shared__ alignas(16) _Float16 Ks[16 * QLD];
  __shared__ alignas(16) _Float16 Vt[64 * VLD];

  const int lane = threadIdx.x;
  const int qb_i = blockIdx.x;           // 0..12
  const int hh   = blockIdx.y;
  const int bb   = blockIdx.z;
  const int q0   = qb_i * 16;
  const size_t bh = (size_t)bb * NHEAD + hh;
  const size_t rowbase = (size_t)bb * TSEQ;

  // ---- stage Q (16 x 64) ----
  for (int c = lane; c < 128; c += 32) {
    int m = c >> 3, off = (c & 7) * 8;
    int t = q0 + m;
    v8h v = {};
    if (t < TSEQ) v = *(const v8h*)(qb + (bh * TSEQ + t) * HDIM + off);
    *(v8h*)(Qs + m * QLD + off) = v;
  }
  __syncthreads();
  v16h aq0 = ld_frag(Qs, QLD, 0, 0,  lane);
  v16h aq1 = ld_frag(Qs, QLD, 0, 32, lane);

  // ---- scores: 13 key blocks of 16 ----
  for (int kblk = 0; kblk < 13; ++kblk) {
    for (int c = lane; c < 128; c += 32) {
      int m = c >> 3, off = (c & 7) * 8;
      int t = kblk * 16 + m;
      v8h v = {};
      if (t < TSEQ) v = *(const v8h*)(kb + (bh * TSEQ + t) * HDIM + off);
      *(v8h*)(Ks + m * QLD + off) = v;
    }
    __syncthreads();
    v8f accs = {};
    accs = wmma_f16(aq0, ld_frag(Ks, QLD, 0, 0,  lane), accs);
    accs = wmma_f16(aq1, ld_frag(Ks, QLD, 0, 32, lane), accs);
#pragma unroll
    for (int r = 0; r < 8; ++r) {
      int rl = r + 8 * (lane >> 4);
      int cl = kblk * 16 + (lane & 15);
      S[rl * SLD + cl] = accs[r] * 8.0f;     // * sqrt(HD) per reference
    }
    __syncthreads();
  }

  // ---- softmax over 197 valid columns; lanes 0..15 own one row each ----
  if (lane < 16) {
    const int r = lane;
    float mx = -1e30f;
    for (int c = 0; c < TSEQ; ++c) mx = fmaxf(mx, S[r * SLD + c]);
    float sum = 0.0f;
    for (int c = 0; c < TSEQ; ++c) {
      float e = expf(S[r * SLD + c] - mx);
      S[r * SLD + c] = e;
      sum += e;
    }
    float inv = 1.0f / sum;
    for (int c = 0; c < TPAD; ++c)
      Ps[r * PLD + c] = (c < TSEQ) ? (_Float16)(S[r * SLD + c] * inv)
                                   : (_Float16)0.0f;
  }
  __syncthreads();

  // ---- O = P @ V : K = 224 (7 steps of 32), N = 64 (4 tiles) ----
  v8f acco[4];
#pragma unroll
  for (int j = 0; j < 4; ++j) acco[j] = {};

  for (int ks = 0; ks < 7; ++ks) {
    int kk0 = ks * 32;
    for (int c = lane; c < 256; c += 32) {   // 64 rows x 4 chunks
      int f = c >> 2, off = (c & 3) * 8;
      *(v8h*)(Vt + f * VLD + off) =
          *(const v8h*)(vb + (bh * HDIM + f) * TPAD + kk0 + off);
    }
    __syncthreads();
    v16h ap = ld_frag(Ps, PLD, 0, kk0, lane);
#pragma unroll
    for (int j = 0; j < 4; ++j)
      acco[j] = wmma_f16(ap, ld_frag(Vt, VLD, j * 16, 0, lane), acco[j]);
    __syncthreads();
  }

#pragma unroll
  for (int j = 0; j < 4; ++j)
#pragma unroll
    for (int r = 0; r < 8; ++r) {
      int t = q0 + r + 8 * (lane >> 4);
      if (t < TSEQ) {
        int f = j * 16 + (lane & 15);
        attn_o[(rowbase + t) * D_MODEL + hh * HDIM + f] = acco[j][r];
      }
    }
}

// ---------------------------------------------------------------------------
// LayerNorm(x + r) * g + b ; one 256-thread block per row of 768
// ---------------------------------------------------------------------------
__global__ __launch_bounds__(256)
void ln_residual(const float* __restrict__ x, const float* __restrict__ r,
                 const float* __restrict__ g, const float* __restrict__ b,
                 float* __restrict__ y) {
  __shared__ float s1[256], s2[256];
  const int row = blockIdx.x;
  const int tid = threadIdx.x;
  const size_t base = (size_t)row * D_MODEL;
  float v[3], sum = 0.0f, sq = 0.0f;
#pragma unroll
  for (int i = 0; i < 3; ++i) {
    int idx = tid + i * 256;
    v[i] = x[base + idx] + r[base + idx];
    sum += v[i];
    sq  += v[i] * v[i];
  }
  s1[tid] = sum; s2[tid] = sq;
  __syncthreads();
  for (int off = 128; off > 0; off >>= 1) {
    if (tid < off) { s1[tid] += s1[tid + off]; s2[tid] += s2[tid + off]; }
    __syncthreads();
  }
  float mean = s1[0] * (1.0f / D_MODEL);
  float var  = s2[0] * (1.0f / D_MODEL) - mean * mean;
  float rstd = rsqrtf(var + EPSLN);
#pragma unroll
  for (int i = 0; i < 3; ++i) {
    int idx = tid + i * 256;
    y[base + idx] = (v[i] - mean) * rstd * g[idx] + b[idx];
  }
}

// ---------------------------------------------------------------------------
// Patchify per reference: x(b,3,224,224) reshape (b,3,P,nx,P,nx)
// -> transpose(0,3,5,2,4,1) -> (b, nx*nx, P*P*3)
// ---------------------------------------------------------------------------
__global__ __launch_bounds__(256)
void patchify(const float* __restrict__ x, float* __restrict__ patches) {
  size_t idx = (size_t)blockIdx.x * 256 + threadIdx.x;
  if (idx >= (size_t)NROWS1 * D_MODEL) return;
  int f   = (int)(idx % D_MODEL);
  int row = (int)(idx / D_MODEL);
  int b   = row / NT;
  int t   = row % NT;
  int r2 = t / NX, c2 = t % NX;
  int ch = f % 3;
  int q  = f / 3;
  int r1 = q / PATCH, c1 = q % PATCH;
  int i = r1 * NX + r2;
  int j = c1 * NX + c2;
  patches[idx] = x[(((size_t)b * 3 + ch) * IMG + i) * IMG + j];
}

__global__ __launch_bounds__(256)
void embed(const float* __restrict__ pproj, const float* __restrict__ proj_b,
           const float* __restrict__ pos, const float* __restrict__ cls,
           float* __restrict__ h) {
  size_t idx = (size_t)blockIdx.x * 256 + threadIdx.x;
  if (idx >= (size_t)NROWS * D_MODEL) return;
  int d   = (int)(idx % D_MODEL);
  int row = (int)(idx / D_MODEL);
  int b   = row / TSEQ;
  int t   = row % TSEQ;
  float v;
  if (t == 0) v = cls[d] + pos[d];
  else v = pproj[((size_t)b * NT + (t - 1)) * D_MODEL + d] + proj_b[d]
           + pos[(size_t)t * D_MODEL + d];
  h[idx] = v;
}

__global__ __launch_bounds__(256)
void extract_tokens(const float* __restrict__ h, float* __restrict__ h1) {
  size_t idx = (size_t)blockIdx.x * 256 + threadIdx.x;
  if (idx >= (size_t)NROWS1 * D_MODEL) return;
  int d   = (int)(idx % D_MODEL);
  int row = (int)(idx / D_MODEL);
  int b   = row / NT;
  int t   = row % NT;
  h1[idx] = h[((size_t)b * TSEQ + 1 + t) * D_MODEL + d];
}

// ---------------------------------------------------------------------------
// host side
// ---------------------------------------------------------------------------
static inline void run_gemm(const float* A, int M, int K, _Float16* a16,
                            const float* B, int N, _Float16* bt,
                            const float* bias, float* C, int act,
                            hipStream_t s) {
  long n8 = (long)M * K / 8;
  cvt_f16<<<dim3((unsigned)((n8 + 255) / 256)), 256, 0, s>>>(A, a16, n8);
  transpose_f16<<<dim3(N / 32, K / 32), 256, 0, s>>>(B, bt, K, N);
  gemm_wmma<<<dim3(N / 128, (M + 127) / 128), 256, 0, s>>>(a16, bt, bias, C,
                                                           M, N, K, act);
}

extern "C" void kernel_launch(void* const* d_in, const int* in_sizes, int n_in,
                              void* d_out, int out_size, void* d_ws, size_t ws_size,
                              hipStream_t stream) {
  const float* x        = (const float*)d_in[0];
  const float* proj_w   = (const float*)d_in[1];
  const float* proj_b   = (const float*)d_in[2];
  const float* pos_emb  = (const float*)d_in[3];
  const float* cls_tok  = (const float*)d_in[4];
  const float* Wqkv     = (const float*)d_in[5];
  const float* Wout     = (const float*)d_in[6];
  const float* ln1_g    = (const float*)d_in[7];
  const float* ln1_b    = (const float*)d_in[8];
  const float* Wm1      = (const float*)d_in[9];
  const float* bm1      = (const float*)d_in[10];
  const float* Wm2      = (const float*)d_in[11];
  const float* bm2      = (const float*)d_in[12];
  const float* ln2_g    = (const float*)d_in[13];
  const float* ln2_b    = (const float*)d_in[14];
  const float* Wo1      = (const float*)d_in[15];
  const float* bo1      = (const float*)d_in[16];
  const float* Wo2      = (const float*)d_in[17];
  const float* bo2      = (const float*)d_in[18];
  float* out = (float*)d_out;

  // carve workspace (floats, 256B aligned).  attn_o and proj_o must stay
  // adjacent: their combined span doubles as the f16 copy of `hidden`.
  char* wp = (char*)d_ws;
  auto carve = [&](size_t nfloat) {
    float* p = (float*)wp;
    size_t bytes = (nfloat * sizeof(float) + 255) & ~(size_t)255;
    wp += bytes;
    return p;
  };
  float* patches = carve((size_t)NROWS1 * D_MODEL);     // later: h1
  float* pproj   = carve((size_t)NROWS1 * D_MODEL);     // later: g1
  float* h       = carve((size_t)NROWS  * D_MODEL);
  float* qkv     = carve((size_t)NROWS  * 3 * D_MODEL); // later: mlp out h2
  float* attn_o  = carve((size_t)NROWS  * D_MODEL);     // + proj_o = big a16
  float* proj_o  = carve((size_t)NROWS  * D_MODEL);
  float* hidden  = carve((size_t)NROWS  * MLPDIM);
  float* btf     = carve((size_t)MLPDIM * D_MODEL / 2); // f16 Bt scratch
  float* h1 = patches;
  float* g1 = pproj;
  float* h2 = qkv;

  _Float16* bt       = (_Float16*)btf;
  _Float16* a16_hid  = (_Float16*)hidden;   // A f16 when hidden is free
  _Float16* a16_att  = (_Float16*)attn_o;   // A f16 when attn_o(+proj_o) free
  const size_t QKSZ = (size_t)BATCH * NHEAD * TSEQ * HDIM;
  const size_t VSZ  = (size_t)BATCH * NHEAD * HDIM * TPAD;
  _Float16* qbuf = (_Float16*)hidden;
  _Float16* kbuf = qbuf + QKSZ;
  _Float16* vbuf = kbuf + QKSZ;

  const size_t nTok  = (size_t)NROWS  * D_MODEL;
  const size_t nTok1 = (size_t)NROWS1 * D_MODEL;

  // --- embedding ---
  patchify<<<dim3((unsigned)((nTok1 + 255) / 256)), 256, 0, stream>>>(x, patches);
  run_gemm(patches, NROWS1, D_MODEL, a16_hid, proj_w, D_MODEL, bt,
           nullptr, pproj, 0, stream);
  embed<<<dim3((unsigned)((nTok + 255) / 256)), 256, 0, stream>>>(
      pproj, proj_b, pos_emb, cls_tok, h);

  // --- transformer layers ---
  for (int l = 0; l < NLAYER; ++l) {
    const float* Wqkv_l = Wqkv + (size_t)l * D_MODEL * 3 * D_MODEL;
    const float* Wout_l = Wout + (size_t)l * D_MODEL * D_MODEL;
    const float* g1_l = ln1_g + (size_t)l * D_MODEL;
    const float* b1_l = ln1_b + (size_t)l * D_MODEL;
    const float* Wm1_l = Wm1 + (size_t)l * D_MODEL * MLPDIM;
    const float* bm1_l = bm1 + (size_t)l * MLPDIM;
    const float* Wm2_l = Wm2 + (size_t)l * MLPDIM * D_MODEL;
    const float* bm2_l = bm2 + (size_t)l * D_MODEL;
    const float* g2_l = ln2_g + (size_t)l * D_MODEL;
    const float* b2_l = ln2_b + (size_t)l * D_MODEL;

    run_gemm(h, NROWS, D_MODEL, a16_hid, Wqkv_l, 3 * D_MODEL, bt,
             nullptr, qkv, 0, stream);
    zero_f16<<<dim3((unsigned)((VSZ / 8 + 255) / 256)), 256, 0, stream>>>(
        vbuf, (long)(VSZ / 8));
    permute_qkv<<<dim3((unsigned)((nTok + 255) / 256)), 256, 0, stream>>>(
        qkv, qbuf, kbuf, vbuf);
    attn_wmma<<<dim3(13, NHEAD, BATCH), 32, 0, stream>>>(qbuf, kbuf, vbuf,
                                                         attn_o);
    run_gemm(attn_o, NROWS, D_MODEL, a16_hid, Wout_l, D_MODEL, bt,
             nullptr, proj_o, 0, stream);
    ln_residual<<<dim3(NROWS), 256, 0, stream>>>(h, proj_o, g1_l, b1_l, h);
    run_gemm(h, NROWS, D_MODEL, a16_att, Wm1_l, MLPDIM, bt,
             bm1_l, hidden, 1, stream);
    run_gemm(hidden, NROWS, MLPDIM, a16_att, Wm2_l, D_MODEL, bt,
             bm2_l, h2, 0, stream);
    ln_residual<<<dim3(NROWS), 256, 0, stream>>>(h, h2, g2_l, b2_l, h);
  }

  // --- head ---
  extract_tokens<<<dim3((unsigned)((nTok1 + 255) / 256)), 256, 0, stream>>>(h, h1);
  run_gemm(h1, NROWS1, D_MODEL, a16_hid, Wo1, OUTC, bt, bo1, g1, 1, stream);
  run_gemm(g1, NROWS1, OUTC, a16_hid, Wo2, OUTC, bt, bo2, out, 0, stream);
  (void)in_sizes; (void)n_in; (void)out_size; (void)ws_size;
}
